// Graphormer_Layer_43198781063735
// MI455X (gfx1250) — matrix operand order repaired and checked by reading, not compiled, not run on hardware
//
#include <hip/hip_runtime.h>
#include <hip/hip_bf16.h>

#define Bb 128
#define Nn 256
#define Ee 1024
#define Dd 128
#define Hh 8
#define DK 16
#define NEGV (-9.0e15f)

typedef __attribute__((ext_vector_type(16))) _Float16     v16h;
typedef __attribute__((ext_vector_type(8)))  float        v8f;
typedef __attribute__((ext_vector_type(4)))  unsigned int v4u;
typedef __attribute__((ext_vector_type(8)))  int          v8i;
typedef __attribute__((ext_vector_type(4)))  int          v4i;

#if __has_builtin(__builtin_amdgcn_tensor_load_to_lds)
#define HAVE_TDM 1
#endif

// K index held by half-slot i (0..15) of lane L for a 16-bit A fragment (16x32, M x K)
// per CDNA5 ISA 7.12.2: VGPR p holds K = {2(p&3),2(p&3)+1} (+8 for lanes>=16, +16 for p>=4)
__device__ __forceinline__ int a16_k(int i, int lane) {
    int p = i >> 1;
    return ((p & 3) << 1) + (i & 1) + ((lane >= 16) ? 8 : 0) + ((p >= 4) ? 16 : 0);
}

#ifdef HAVE_TDM
// Issue a 1-D TDM tensor_load_to_lds: nelem2B 2-byte elements (2*nelem2B bytes, multiple of 4) from gaddr -> LDS ldsaddr.
__device__ __forceinline__ void tdm_load_1d(unsigned ldsaddr, unsigned long long gaddr,
                                            unsigned nelem2B) {
    v4u g0;
    g0[0] = 1u;                                          // count=1, user descriptor
    g0[1] = ldsaddr;                                     // lds_addr
    g0[2] = (unsigned)(gaddr & 0xFFFFFFFFull);           // global_addr[31:0]
    g0[3] = (unsigned)((gaddr >> 32) & 0x1FFFFFFull) | (2u << 30);  // addr[56:32] | type=2
    v8i g1;
    g1[0] = (int)(1u << 16);                             // data_size=1 (2 bytes)
    g1[1] = (int)(nelem2B << 16);                        // tensor_dim0[15:0]
    g1[2] = (int)(1u << 16);                             // tensor_dim0[31:16]=0 | tensor_dim1=1
    g1[3] = (int)(nelem2B << 16);                        // tile_dim0
    g1[4] = 1;                                           // tile_dim1=1
    g1[5] = (int)nelem2B;                                // tensor_dim0_stride[31:0]
    g1[6] = 0; g1[7] = 0;
    v4i g2 = {}; v4i g3 = {};
#if defined(__clang_major__) && (__clang_major__ >= 23)
    v8i g4 = {};
    __builtin_amdgcn_tensor_load_to_lds(g0, g1, g2, g3, g4, 0);
#else
    __builtin_amdgcn_tensor_load_to_lds(g0, g1, g2, g3, 0);
#endif
}
#endif

// ---------------- kernel 0: column-mean of We_w, mean of We_b -----------------
__global__ void wred_kernel(const float* __restrict__ We_w, const float* __restrict__ We_b,
                            float* __restrict__ wred, float* __restrict__ bmean) {
    int i = threadIdx.x;            // 128 threads
    float s = 0.f;
    for (int j = 0; j < Dd; ++j) s += We_w[i * Dd + j];
    wred[i] = s * (1.f / Dd);
    if (i == 0) {
        float sb = 0.f;
        for (int j = 0; j < Dd; ++j) sb += We_b[j];
        *bmean = sb * (1.f / Dd);
    }
}

// ---------------- kernel 0b: pack Wq/Wk/Wv rows 0..127 into f16 B-fragment order
// layout per matrix: pk[kc*4096 + h*2048 + j*16 + i] = W[(kc*32 + 8h + i + (i>=8 ? 8 : 0))*128 + j]
__global__ void wpack_kernel(const float* __restrict__ Wq, const float* __restrict__ Wk,
                             const float* __restrict__ Wv, _Float16* __restrict__ pk) {
    const int idx = blockIdx.x * 256 + threadIdx.x;      // 3*16384 threads
    if (idx >= 3 * 16384) return;
    const float* W = (idx < 16384) ? Wq : (idx < 32768 ? Wk : Wv);
    const int r  = idx & 16383;
    const int i  = r & 15;
    const int j  = (r >> 4) & 127;
    const int hh = (r >> 11) & 1;
    const int kc = r >> 12;
    pk[idx] = (_Float16)W[(kc * 32 + 8 * hh + i + ((i >= 8) ? 8 : 0)) * Dd + j];
}

// ---------------- kernel 1: degrees, bias scatter, x -> f16 A-fragment order --
__global__ void prep_kernel(const float* __restrict__ x, const int* __restrict__ edge_index,
                            const float* __restrict__ edge_attr,
                            const float* __restrict__ wred, const float* __restrict__ bmean,
                            float* __restrict__ degr, float* __restrict__ degc,
                            float* __restrict__ bias, _Float16* __restrict__ xpA) {
    __shared__ float sdr[Nn], sdc[Nn], swred[Dd];
    const int b = blockIdx.x, tid = threadIdx.x;      // 256 threads
    sdr[tid] = 0.f; sdc[tid] = 0.f;
    if (tid < Dd) swred[tid] = wred[tid];

    const int* rows = edge_index + (size_t)b * 2 * Ee;
    const int* cols = rows + Ee;
    float* bb = bias + (size_t)b * Nn * Nn;
    for (int i = tid; i < Nn * Nn / 4; i += 256)
        ((float4*)bb)[i] = make_float4(0.f, 0.f, 0.f, 0.f);

    // x -> f16 in A-fragment order: xpA[((b*16+mt)*4+kc)*512 + lane*16 + i]
    const float* xb = x + (size_t)b * Nn * Dd;
    _Float16* xp = xpA + (size_t)b * Nn * Dd;
    for (int idx = tid; idx < Nn * Dd; idx += 256) {
        const int row = idx >> 7, col = idx & 127;
        const int kc = col >> 5, kl = col & 31;
        const int Lhi = (kl >> 3) & 1;
        const int p = ((kl >> 4) & 1) * 4 + ((kl & 7) >> 1);
        const int i = 2 * p + (kl & 1);
        const int lane = (row & 15) + (Lhi << 4);
        xp[(((row >> 4) * 4 + kc) << 9) + lane * 16 + i] = (_Float16)xb[idx];
    }
    __syncthreads();
    for (int e = tid; e < Ee; e += 256) {
        atomicAdd(&sdr[rows[e]], 1.0f);
        atomicAdd(&sdc[cols[e]], 1.0f);
    }
    __syncthreads();
    degr[b * Nn + tid] = sdr[tid];
    degc[b * Nn + tid] = sdc[tid];
    // edge feature: mean(A@W + b) == A . colmean(W) + mean(b)
    const float bm = *bmean;
    for (int e = tid; e < Ee; e += 256) {
        const float* ea = edge_attr + ((size_t)b * Ee + e) * Dd;
        float f = bm;
        for (int j = 0; j < Dd; ++j) f += ea[j] * swred[j];
        bb[rows[e] * Nn + cols[e]] = f;
    }
}

// ---------------- kernel 2: QKV projection via WMMA ---------------------------
// grid = B * 16 m-tiles, block = 256 (8 waves; wave w == head w == output cols [16w,16w+16))
__global__ void qkv_kernel(const _Float16* __restrict__ xpA,
                           const _Float16* __restrict__ wpk,
                           const float* __restrict__ Wq, const float* __restrict__ Wk,
                           const float* __restrict__ Wv,
                           const float* __restrict__ degr, const float* __restrict__ degc,
                           _Float16* __restrict__ q16, _Float16* __restrict__ k16,
                           _Float16* __restrict__ v16) {
    const int b    = blockIdx.x >> 4;
    const int mt   = blockIdx.x & 15;
    const int wave = threadIdx.x >> 5;     // head
    const int lane = threadIdx.x & 31;
    const int col0 = wave * DK;
    const int n    = lane & 15;

    const _Float16* xa  = xpA + (size_t)b * Nn * Dd + (mt * 4) * 512 + lane * 16;
    const _Float16* wqp = wpk;
    const _Float16* wkp = wpk + 16384;
    const _Float16* wvp = wpk + 32768;
    const int fj = ((lane >> 4) * 128 + col0 + n) * 16;  // fragment offset within kc slice

    v8f accq = {}, acck = {}, accv = {};
#pragma unroll
    for (int kc = 0; kc < 4; ++kc) {       // K = 128 in 4 chunks of 32
        const v16h a  = *(const v16h*)(xa + (kc << 9));
        const v16h bq = *(const v16h*)(wqp + kc * 4096 + fj);
        const v16h bk = *(const v16h*)(wkp + kc * 4096 + fj);
        const v16h bv = *(const v16h*)(wvp + kc * 4096 + fj);
        accq = __builtin_amdgcn_wmma_f32_16x16x32_f16(false, a, false, bq, (short)0, accq, false, false);
        acck = __builtin_amdgcn_wmma_f32_16x16x32_f16(false, a, false, bk, (short)0, acck, false, false);
        accv = __builtin_amdgcn_wmma_f32_16x16x32_f16(false, a, false, bv, (short)0, accv, false, false);
    }
    // rank-2 fp32 update for the two degree feature rows (W rows 128,129)
    const float wq0 = Wq[128 * Dd + col0 + n], wq1 = Wq[129 * Dd + col0 + n];
    const float wk0 = Wk[128 * Dd + col0 + n], wk1 = Wk[129 * Dd + col0 + n];
    const float wv0 = Wv[128 * Dd + col0 + n], wv1 = Wv[129 * Dd + col0 + n];
    const size_t base = ((size_t)b * Hh + wave) * Nn * DK;
#pragma unroll
    for (int r = 0; r < 8; ++r) {
        const int m  = mt * 16 + r + ((lane >= 16) ? 8 : 0);
        const float dr = degr[b * Nn + m], dc = degc[b * Nn + m];
        const float q = accq[r] + dr * wq0 + dc * wq1;
        const float k = acck[r] + dr * wk0 + dc * wk1;
        const float v = accv[r] + dr * wv0 + dc * wv1;
        q16[base + m * DK + n] = (_Float16)(q * 0.25f);   // fold 1/sqrt(DK)
        k16[base + m * DK + n] = (_Float16)k;
        v16[base + m * DK + n] = (_Float16)v;
    }
}

// ---------------- kernel 3: fused attention per (b,h) -------------------------
// grid = B*H, block = 256 (8 waves * 32 rows). Dynamic LDS: S (256KB) + K,V (16KB) + rowsum (1KB)
__global__ void attn_kernel(const _Float16* __restrict__ q16, const _Float16* __restrict__ k16,
                            const _Float16* __restrict__ v16, const float* __restrict__ bias,
                            const unsigned char* __restrict__ mask, float* __restrict__ O) {
    extern __shared__ char smem_raw[];
    float*    S      = (float*)smem_raw;                                  // 256*256 f32
    _Float16* Kh     = (_Float16*)(smem_raw + Nn * Nn * 4);               // 256*16 f16
    _Float16* Vh     = (_Float16*)(smem_raw + Nn * Nn * 4 + Nn * DK * 2); // 256*16 f16
    float*    rowsum = (float*)(smem_raw + Nn * Nn * 4 + 2 * Nn * DK * 2);

    const int b = blockIdx.x >> 3;
    const int h = blockIdx.x & 7;
    const int tid = threadIdx.x, wave = tid >> 5, lane = tid & 31;
    const int n = lane & 15;
    const size_t hb = ((size_t)b * Hh + h) * Nn * DK;

#ifdef HAVE_TDM
    // Stage K and V tiles (8KB each) into LDS with the Tensor Data Mover.
    if (wave == 0) {
        tdm_load_1d((unsigned)(size_t)(void*)Kh, (unsigned long long)(size_t)(k16 + hb), Nn * DK);
        __builtin_amdgcn_s_wait_tensorcnt(0);
    } else if (wave == 1) {
        tdm_load_1d((unsigned)(size_t)(void*)Vh, (unsigned long long)(size_t)(v16 + hb), Nn * DK);
        __builtin_amdgcn_s_wait_tensorcnt(0);
    }
#else
    for (int i = tid; i < Nn * DK; i += 256) { Kh[i] = k16[hb + i]; Vh[i] = v16[hb + i]; }
#endif
    __builtin_prefetch(bias + ((size_t)b * Nn + tid) * Nn, 0, 1);
    __syncthreads();

    // ---- S = (Q/sqrt(dk)) . K^T via WMMA (K-dim 16, zero-padded to 32) ----
    for (int mt2 = 0; mt2 < 2; ++mt2) {
        const int mt = wave * 2 + mt2;
        v16h a;
        const _Float16* qrow = q16 + hb + (mt * 16 + (lane & 15)) * DK;
#pragma unroll
        for (int i = 0; i < 16; ++i) {
            const int k = a16_k(i, lane);
            a[i] = (k < DK) ? qrow[k] : (_Float16)0.0f;
        }
        for (int nt = 0; nt < 16; ++nt) {
            v16h bfrag;
#pragma unroll
            for (int i = 0; i < 16; ++i) {
                const int k = a16_k(i, lane);
                bfrag[i] = (k < DK) ? Kh[(nt * 16 + n) * DK + k] : (_Float16)0.0f;
            }
            v8f c = {};
            c = __builtin_amdgcn_wmma_f32_16x16x32_f16(false, a, false, bfrag, (short)0, c, false, false);
#pragma unroll
            for (int r = 0; r < 8; ++r) {
                const int m = mt * 16 + r + ((lane >= 16) ? 8 : 0);
                S[m * Nn + nt * 16 + n] = c[r];
            }
        }
    }
    __syncthreads();

    // ---- masked bias + softmax (row per thread); keep un-normalized exp in S ----
    {
        const int row = tid;
        const float* brow = bias + ((size_t)b * Nn + row) * Nn;
        float mx = -3.0e38f;
        for (int j = 0; j < Nn; ++j) {
            float s = S[row * Nn + j];
            s = (mask[b * Nn + j] ? s : NEGV) + brow[j];
            S[row * Nn + j] = s;
            mx = fmaxf(mx, s);
        }
        float sum = 0.f;
        for (int j = 0; j < Nn; ++j) {
            const float e = __expf(S[row * Nn + j] - mx);
            S[row * Nn + j] = e;
            sum += e;
        }
        rowsum[row] = sum;
    }
    __syncthreads();

    // ---- O = P . V via WMMA (K = 256 in 8 chunks of 32); normalize on store ----
    for (int mt2 = 0; mt2 < 2; ++mt2) {
        const int mt = wave * 2 + mt2;
        v8f c = {};
        const int arow = mt * 16 + (lane & 15);
        for (int kc = 0; kc < 8; ++kc) {
            v16h a, bfrag;
#pragma unroll
            for (int i = 0; i < 16; ++i) a[i] = (_Float16)S[arow * Nn + kc * 32 + a16_k(i, lane)];
            const int kb = kc * 32 + ((lane >= 16) ? 16 : 0);
#pragma unroll
            for (int i = 0; i < 16; ++i) bfrag[i] = Vh[(kc * 32 + a16_k(i, lane)) * DK + n];
            c = __builtin_amdgcn_wmma_f32_16x16x32_f16(false, a, false, bfrag, (short)0, c, false, false);
        }
#pragma unroll
        for (int r = 0; r < 8; ++r) {
            const int m = mt * 16 + r + ((lane >= 16) ? 8 : 0);
            O[((size_t)b * Nn + m) * Dd + h * DK + n] = c[r] / rowsum[m];
        }
    }
}

// ---------------- kernel 4: residual + LayerNorm ------------------------------
__global__ void resid_ln_kernel(const float* __restrict__ O, const float* __restrict__ x,
                                const float* __restrict__ g, const float* __restrict__ bt,
                                float* __restrict__ out) {
    const int row = blockIdx.x * blockDim.x + threadIdx.x;   // B*N rows
    if (row >= Bb * Nn) return;
    const float* o  = O + (size_t)row * Dd;
    const float* xr = x + (size_t)row * Dd;
    float s1 = 0.f, s2 = 0.f;
    for (int j = 0; j < Dd; ++j) { const float v = o[j] + xr[j]; s1 += v; s2 += v * v; }
    const float mean = s1 * (1.f / Dd);
    const float var  = s2 * (1.f / Dd) - mean * mean;
    const float inv  = rsqrtf(var + 1e-6f);
    float* outr = out + (size_t)row * Dd;
    for (int j = 0; j < Dd; ++j) {
        const float v = o[j] + xr[j];
        outr[j] = (v - mean) * inv * g[j] + bt[j];
    }
}

extern "C" void kernel_launch(void* const* d_in, const int* in_sizes, int n_in,
                              void* d_out, int out_size, void* d_ws, size_t ws_size,
                              hipStream_t stream) {
    const float*         x          = (const float*)d_in[0];
    const unsigned char* mask       = (const unsigned char*)d_in[1];
    const int*           edge_index = (const int*)d_in[2];
    const float*         edge_attr  = (const float*)d_in[3];
    const float*         Wq         = (const float*)d_in[4];
    const float*         Wk         = (const float*)d_in[5];
    const float*         Wv         = (const float*)d_in[6];
    const float*         We_w       = (const float*)d_in[7];
    const float*         We_b       = (const float*)d_in[8];
    const float*         ln_g       = (const float*)d_in[9];
    const float*         ln_b       = (const float*)d_in[10];

    char* ws = (char*)d_ws;
    size_t off = 0;
    float* wred  = (float*)(ws + off); off += 512;
    float* bmean = (float*)(ws + off); off += 512;
    float* degr  = (float*)(ws + off); off += (size_t)Bb * Nn * 4;
    float* degc  = (float*)(ws + off); off += (size_t)Bb * Nn * 4;
    _Float16* wpk = (_Float16*)(ws + off); off += 3 * 16384 * 2;
    float* biasb = (float*)(ws + off); off += (size_t)Bb * Nn * Nn * 4;
    _Float16* xpA = (_Float16*)(ws + off); off += (size_t)Bb * Nn * Dd * 2;
    _Float16* q16 = (_Float16*)(ws + off); off += (size_t)Bb * Nn * Dd * 2;
    _Float16* k16 = (_Float16*)(ws + off); off += (size_t)Bb * Nn * Dd * 2;
    _Float16* v16 = (_Float16*)(ws + off); off += (size_t)Bb * Nn * Dd * 2;
    float* Obuf   = (float*)(ws + off);    off += (size_t)Bb * Nn * Dd * 4;

    wred_kernel<<<1, 128, 0, stream>>>(We_w, We_b, wred, bmean);
    wpack_kernel<<<(3 * 16384 + 255) / 256, 256, 0, stream>>>(Wq, Wk, Wv, wpk);
    prep_kernel<<<Bb, 256, 0, stream>>>(x, edge_index, edge_attr, wred, bmean,
                                        degr, degc, biasb, xpA);
    qkv_kernel<<<Bb * 16, 256, 0, stream>>>(xpA, wpk, Wq, Wk, Wv, degr, degc, q16, k16, v16);
    const size_t smem = (size_t)Nn * Nn * 4 + 2 * (size_t)Nn * DK * 2 + Nn * 4;
    attn_kernel<<<Bb * Hh, 256, smem, stream>>>(q16, k16, v16, biasb, mask, Obuf);
    resid_ln_kernel<<<(Bb * Nn + 255) / 256, 256, 0, stream>>>(Obuf, x, ln_g, ln_b, (float*)d_out);
}
